// MathExpert_34488587387267
// MI455X (gfx1250) — matrix operand order repaired and checked by reading, not compiled, not run on hardware
//
#include <hip/hip_runtime.h>

typedef __attribute__((ext_vector_type(16))) __bf16 v16bf;
typedef __attribute__((ext_vector_type(8)))  __bf16 v8bf;
typedef __attribute__((ext_vector_type(8)))  float  v8f;

#define BATCH   16384
#define IN_DIM  1024
#define HID     4096
#define H2      2048
#define H4      1024

#define KSTEP   32
#define LSTR    40    // padded LDS row stride (elements): (n*20)%64 distinct -> no bank conflicts

// ---------------------------------------------------------------------------
// CDNA5 async global->LDS DMA (ASYNCcnt tracked), bypasses VGPRs
__device__ inline void async_b128(const void* g, unsigned ldsoff) {
  asm volatile("global_load_async_to_lds_b128 %0, %1, off"
               :: "v"(ldsoff), "v"(g) : "memory");
}
__device__ inline void wait_async0() {
  asm volatile("s_wait_asynccnt 0" ::: "memory");
}

// ---------------------------------------------------------------------------
// f32 -> bf16 elementwise convert
__global__ __launch_bounds__(256) void k_cvt_bf16(const float* __restrict__ in,
                                                  __bf16* __restrict__ out, int n) {
  int i = blockIdx.x * 256 + threadIdx.x;
  if (i < n) out[i] = (__bf16)in[i];
}

// W (K x N, f32, row major) -> Wt (N x K, bf16, row major); 32x32 LDS-tiled
__global__ __launch_bounds__(256) void k_wtrans(const float* __restrict__ W,
                                                __bf16* __restrict__ Wt, int K, int N) {
  __shared__ float t[32][33];
  int n0 = blockIdx.x * 32, k0 = blockIdx.y * 32;
  int tx = threadIdx.x & 31, ty = threadIdx.x >> 5;   // 32 x 8
#pragma unroll
  for (int i = 0; i < 32; i += 8)
    t[ty + i][tx] = W[(size_t)(k0 + ty + i) * N + n0 + tx];     // coalesced in n
  __syncthreads();
#pragma unroll
  for (int i = 0; i < 32; i += 8)
    Wt[(size_t)(n0 + ty + i) * K + k0 + tx] = (__bf16)t[tx][ty + i]; // coalesced in k
}

// ---------------------------------------------------------------------------
// WMMA GEMM:  C(MxN) = A(MxK,bf16) * Wt(NxK,bf16)^T + bias
// block = 256 thr (8 waves), 128x128 tile; each wave computes 32x64.
// Double-buffered async global->LDS staging, one barrier per K-step.
// mode bit0: relu, bit1: store f32 (else bf16)
__device__ inline v16bf frag2(const __bf16* p0, const __bf16* p1) {
  union { v16bf v; v8bf h[2]; } u;
  u.h[0] = *(const v8bf*)p0;
  u.h[1] = *(const v8bf*)p1;
  return u.v;
}

__global__ __launch_bounds__(256) void k_gemm_wmma(
    const __bf16* __restrict__ A, const __bf16* __restrict__ Wt,
    const float* __restrict__ bias, void* __restrict__ out,
    int M, int N, int K, int mode) {
  const int tid  = threadIdx.x;
  const int lane = tid & 31;
  const int wave = tid >> 5;
  const int wm   = wave >> 1;          // 0..3 : 32-row strip
  const int wn   = wave & 1;           // 0..1 : 64-col strip
  const int m0 = blockIdx.y * 128;
  const int n0 = blockIdx.x * 128;

  __shared__ __attribute__((aligned(16))) __bf16 ldsA[2][128 * LSTR];
  __shared__ __attribute__((aligned(16))) __bf16 ldsB[2][128 * LSTR];

  v8f acc[8] = {};

  // staging: 128 rows x 32 elems = 512 16B-chunks per matrix, 2 per thread
  const int sr = tid >> 2;             // 0..63 (rows sr and sr+64)
  const int sc = (tid & 3) * 8;        // element col 0,8,16,24
  const __bf16* Ag = A  + (size_t)(m0 + sr) * K + sc;
  const __bf16* Bg = Wt + (size_t)(n0 + sr) * K + sc;
  const unsigned lds_thr = (unsigned)(sr * LSTR + sc) * 2;
  unsigned aoff[2], boff[2];
  aoff[0] = (unsigned)(uintptr_t)&ldsA[0][0] + lds_thr;
  aoff[1] = (unsigned)(uintptr_t)&ldsA[1][0] + lds_thr;
  boff[0] = (unsigned)(uintptr_t)&ldsB[0][0] + lds_thr;
  boff[1] = (unsigned)(uintptr_t)&ldsB[1][0] + lds_thr;

  auto stage = [&](int kt, int buf) {
    const __bf16* ga = Ag + (size_t)kt * KSTEP;
    const __bf16* gb = Bg + (size_t)kt * KSTEP;
    unsigned la = aoff[buf], lb = boff[buf];
#pragma unroll
    for (int i = 0; i < 2; ++i) {
      async_b128(ga, la);
      async_b128(gb, lb);
      ga += (size_t)64 * K;   gb += (size_t)64 * K;
      la += 64 * LSTR * 2;    lb += 64 * LSTR * 2;
    }
  };

  const int am0 = (wm * 32 + (lane & 15)) * LSTR;   // A row base (elements)
  const int ak  = (lane >> 4) * 8;                  // A k sub-offset
  const int bk  = (lane >> 4) * 8;                  // B k sub-offset

  const int nk = K / KSTEP;
  stage(0, 0);
  int cur = 0;
  for (int kt = 0; kt < nk; ++kt) {
    wait_async0();            // tile kt landed in LDS (this wave's chunks)
    __syncthreads();          // everyone's chunks visible; prev buffer fully read
    if (kt + 1 < nk) stage(kt + 1, cur ^ 1);   // overlap DMA with compute
    const __bf16* la = &ldsA[cur][0];
    const __bf16* lb = &ldsB[cur][0];
    v16bf af0 = frag2(la + am0 + ak,             la + am0 + ak + 16);
    v16bf af1 = frag2(la + am0 + 16 * LSTR + ak, la + am0 + 16 * LSTR + ak + 16);
#pragma unroll
    for (int ni = 0; ni < 4; ++ni) {
      int bn = (wn * 64 + ni * 16 + (lane & 15)) * LSTR;
      v16bf bf = frag2(lb + bn + bk, lb + bn + bk + 16);
      acc[ni]     = __builtin_amdgcn_wmma_f32_16x16x32_bf16(
          false, af0, false, bf, (short)0, acc[ni], false, false);
      acc[4 + ni] = __builtin_amdgcn_wmma_f32_16x16x32_bf16(
          false, af1, false, bf, (short)0, acc[4 + ni], false, false);
    }
    cur ^= 1;
  }

  // epilogue: acc[mi*4+ni][i] -> row m0+wm*32+mi*16+8*(lane>>4)+i, col n0+wn*64+ni*16+(lane&15)
#pragma unroll
  for (int mi = 0; mi < 2; ++mi) {
#pragma unroll
    for (int ni = 0; ni < 4; ++ni) {
      int n = n0 + wn * 64 + ni * 16 + (lane & 15);
      float bv = bias[n];
#pragma unroll
      for (int i = 0; i < 8; ++i) {
        int m = m0 + wm * 32 + mi * 16 + (lane >> 4) * 8 + i;
        float v = acc[mi * 4 + ni][i] + bv;
        if (mode & 1) v = fmaxf(v, 0.0f);
        if (mode & 2) ((float*) out)[(size_t)m * N + n] = v;
        else          ((__bf16*)out)[(size_t)m * N + n] = (__bf16)v;
      }
    }
  }
}

// ---------------------------------------------------------------------------
// LayerNorm (+gamma/beta) + ReLU over rows of length C; bf16 in -> bf16 out
__global__ __launch_bounds__(256) void k_ln_relu(const __bf16* __restrict__ in,
                                                 const float* __restrict__ g,
                                                 const float* __restrict__ beta,
                                                 __bf16* __restrict__ out, int C) {
  int row = blockIdx.x, tid = threadIdx.x;
  const __bf16* x = in + (size_t)row * C;
  float s = 0.f, ss = 0.f;
  for (int c = tid * 8; c < C; c += 2048) {
    union { uint4 u; __bf16 b[8]; } v; v.u = *(const uint4*)&x[c];
#pragma unroll
    for (int j = 0; j < 8; ++j) { float f = (float)v.b[j]; s += f; ss += f * f; }
  }
  __shared__ float r0[256], r1[256];
  r0[tid] = s; r1[tid] = ss;
  __syncthreads();
  for (int st = 128; st > 0; st >>= 1) {
    if (tid < st) { r0[tid] += r0[tid + st]; r1[tid] += r1[tid + st]; }
    __syncthreads();
  }
  float mu = r0[0] / C;
  float var = r1[0] / C - mu * mu;
  float rs = rsqrtf(var + 1e-5f);
  for (int c = tid * 8; c < C; c += 2048) {
    union { uint4 u; __bf16 b[8]; } v; v.u = *(const uint4*)&x[c];
    union { uint4 u; __bf16 b[8]; } o;
#pragma unroll
    for (int j = 0; j < 8; ++j) {
      float f = ((float)v.b[j] - mu) * rs * g[c + j] + beta[c + j];
      o.b[j] = (__bf16)fmaxf(f, 0.0f);
    }
    *(uint4*)&out[(size_t)row * C + c] = o.u;
  }
}

// ---------------------------------------------------------------------------
// head: logits[j] = sum_c H[row,c]*W2[c,j] + b2[j]   (nout<=4)
__global__ __launch_bounds__(256) void k_head(const __bf16* __restrict__ H,
                                              const float* __restrict__ W2,
                                              const float* __restrict__ b2,
                                              int C, int nout, int is_od,
                                              int* __restrict__ op_out,
                                              float* __restrict__ num_out,
                                              float* __restrict__ dout) {
  int row = blockIdx.x, tid = threadIdx.x;
  const __bf16* h = H + (size_t)row * C;
  float acc[4] = {0.f, 0.f, 0.f, 0.f};
  for (int c = tid; c < C; c += 256) {
    float hv = (float)h[c];
    for (int j = 0; j < nout; ++j) acc[j] += hv * W2[c * nout + j];
  }
  __shared__ float red[4][256];
  for (int j = 0; j < nout; ++j) red[j][tid] = acc[j];
  __syncthreads();
  for (int st = 128; st > 0; st >>= 1) {
    if (tid < st)
      for (int j = 0; j < nout; ++j) red[j][tid] += red[j][tid + st];
    __syncthreads();
  }
  if (tid == 0) {
    if (is_od) {
      int best = 0; float bv = red[0][0] + b2[0];
      for (int j = 1; j < nout; ++j) {
        float v = red[j][0] + b2[j];
        if (v > bv) { bv = v; best = j; }
      }
      op_out[row] = best;
      dout[row] = (float)best;
    } else {
      for (int j = 0; j < nout; ++j) {
        float v = red[j][0] + b2[j];
        num_out[row * 2 + j] = v;
        dout[row * 2 + j] = v;
      }
    }
  }
}

// ---------------------------------------------------------------------------
// selected-expert MLP + neural/symbolic outputs (block per row)
__global__ __launch_bounds__(256) void k_expert(const float* __restrict__ numbers,
                                                const int* __restrict__ op,
                                                const float* __restrict__ op_w1,
                                                const float* __restrict__ op_b1,
                                                const float* __restrict__ op_w2,
                                                const float* __restrict__ op_b2,
                                                float* __restrict__ neural_buf,
                                                float* __restrict__ dout_neural,
                                                float* __restrict__ dout_sym) {
  int row = blockIdx.x, tid = threadIdx.x;
  float a = numbers[row * 2 + 0];
  float b = numbers[row * 2 + 1];
  int e = op[row];
  const float* w1a = op_w1 + (size_t)e * 2 * H4;
  const float* w1b = w1a + H4;
  const float* b1  = op_b1 + (size_t)e * H4;
  const float* w2  = op_w2 + (size_t)e * H4;
  float part = 0.f;
  for (int h = tid; h < H4; h += 256) {
    float eh = fmaxf(a * w1a[h] + b * w1b[h] + b1[h], 0.0f);
    part += eh * w2[h];
  }
  __shared__ float red[256];
  red[tid] = part;
  __syncthreads();
  for (int st = 128; st > 0; st >>= 1) {
    if (tid < st) red[tid] += red[tid + st];
    __syncthreads();
  }
  if (tid == 0) {
    float neural = red[0] + op_b2[e];
    if (e == 3 && b == 0.0f) neural = __builtin_nanf("");
    neural_buf[row] = neural;
    dout_neural[row] = neural;
    float sym;
    if      (e == 0) sym = a + b;
    else if (e == 1) sym = a - b;
    else if (e == 2) sym = a * b;
    else             sym = (b == 0.0f) ? __builtin_nanf("") : a / b;
    dout_sym[row] = sym;
  }
}

// r1[m,j] = relu(neural[m]*re_w1[j] + re_b1[j]) -> bf16
__global__ __launch_bounds__(256) void k_rank1(const float* __restrict__ neural,
                                               const float* __restrict__ w1,
                                               const float* __restrict__ b1,
                                               __bf16* __restrict__ out) {
  int i = blockIdx.x * 256 + threadIdx.x;
  int m = i >> 10;           // /1024
  int j = i & 1023;
  float v = fmaxf(neural[m] * w1[j] + b1[j], 0.0f);
  out[i] = (__bf16)v;
}

// ---------------------------------------------------------------------------
extern "C" void kernel_launch(void* const* d_in, const int* in_sizes, int n_in,
                              void* d_out, int out_size, void* d_ws, size_t ws_size,
                              hipStream_t stream) {
  const float* x      = (const float*)d_in[0];
  const float* enc_w1 = (const float*)d_in[1];
  const float* enc_b1 = (const float*)d_in[2];
  const float* ln1_g  = (const float*)d_in[3];
  const float* ln1_b  = (const float*)d_in[4];
  const float* enc_w2 = (const float*)d_in[5];
  const float* enc_b2 = (const float*)d_in[6];
  const float* ln2_g  = (const float*)d_in[7];
  const float* ln2_b  = (const float*)d_in[8];
  const float* od_w1  = (const float*)d_in[9];
  const float* od_b1  = (const float*)d_in[10];
  const float* od_w2  = (const float*)d_in[11];
  const float* od_b2  = (const float*)d_in[12];
  const float* ne_w1  = (const float*)d_in[13];
  const float* ne_b1  = (const float*)d_in[14];
  const float* ne_w2  = (const float*)d_in[15];
  const float* ne_b2  = (const float*)d_in[16];
  const float* op_w1  = (const float*)d_in[17];
  const float* op_b1  = (const float*)d_in[18];
  const float* op_w2  = (const float*)d_in[19];
  const float* op_b2  = (const float*)d_in[20];
  const float* re_w1  = (const float*)d_in[21];
  const float* re_b1  = (const float*)d_in[22];
  const float* re_w2  = (const float*)d_in[23];
  const float* re_b2  = (const float*)d_in[24];
  const float* re_w3  = (const float*)d_in[25];
  const float* re_b3  = (const float*)d_in[26];

  // workspace layout
  __bf16* w1t  = (__bf16*)d_ws;                         // 4096x1024
  __bf16* w2t  = w1t  + (size_t)HID * IN_DIM;           // 4096x4096
  __bf16* od1t = w2t  + (size_t)HID * HID;              // 2048x4096
  __bf16* ne1t = od1t + (size_t)H2  * HID;              // 2048x4096
  __bf16* re2t = ne1t + (size_t)H2  * HID;              // 2048x1024
  __bf16* re3t = re2t + (size_t)H2  * H4;               // 1024x2048
  __bf16* bufA = re3t + (size_t)IN_DIM * H2;            // B x 4096 bf16
  __bf16* bufB = bufA + (size_t)BATCH * HID;            // B x 4096 bf16
  int*    opb  = (int*)  (bufB + (size_t)BATCH * HID);  // B ints
  float*  numb = (float*)(opb + BATCH);                 // B x 2
  float*  neub = numb + (size_t)BATCH * 2;              // B

  float* out_repr   = (float*)d_out;                    // B x 1024
  float* out_op     = out_repr + (size_t)BATCH * IN_DIM;
  float* out_num    = out_op + BATCH;
  float* out_neural = out_num + (size_t)BATCH * 2;
  float* out_sym    = out_neural + BATCH;

  // --- weight conversions (deterministic, every call) ---
  auto nblk = [](size_t n) { return (unsigned)((n + 255) / 256); };
  k_wtrans<<<dim3(HID / 32, IN_DIM / 32), 256, 0, stream>>>(enc_w1, w1t, IN_DIM, HID);
  k_wtrans<<<dim3(HID / 32, HID / 32),    256, 0, stream>>>(enc_w2, w2t, HID, HID);
  k_wtrans<<<dim3(H2 / 32, HID / 32),     256, 0, stream>>>(od_w1, od1t, HID, H2);
  k_wtrans<<<dim3(H2 / 32, HID / 32),     256, 0, stream>>>(ne_w1, ne1t, HID, H2);
  k_wtrans<<<dim3(H2 / 32, H4 / 32),      256, 0, stream>>>(re_w2, re2t, H4, H2);
  k_wtrans<<<dim3(IN_DIM / 32, H2 / 32),  256, 0, stream>>>(re_w3, re3t, H2, IN_DIM);
  k_cvt_bf16<<<nblk((size_t)BATCH * IN_DIM), 256, 0, stream>>>(x, bufA, BATCH * IN_DIM);

  // --- encoder ---
  dim3 blk(256);
  k_gemm_wmma<<<dim3(HID / 128, BATCH / 128), blk, 0, stream>>>(
      bufA, w1t, enc_b1, bufB, BATCH, HID, IN_DIM, /*mode=*/0);
  k_ln_relu<<<BATCH, 256, 0, stream>>>(bufB, ln1_g, ln1_b, bufA, HID);
  k_gemm_wmma<<<dim3(HID / 128, BATCH / 128), blk, 0, stream>>>(
      bufA, w2t, enc_b2, bufB, BATCH, HID, HID, /*mode=*/0);
  k_ln_relu<<<BATCH, 256, 0, stream>>>(bufB, ln2_g, ln2_b, bufA, HID);   // h in bufA

  // --- operation head ---
  k_gemm_wmma<<<dim3(H2 / 128, BATCH / 128), blk, 0, stream>>>(
      bufA, od1t, od_b1, bufB, BATCH, H2, HID, /*mode=*/1);              // relu, bf16
  k_head<<<BATCH, 256, 0, stream>>>(bufB, od_w2, od_b2, H2, 4, 1, opb, numb, out_op);

  // --- numbers head ---
  k_gemm_wmma<<<dim3(H2 / 128, BATCH / 128), blk, 0, stream>>>(
      bufA, ne1t, ne_b1, bufB, BATCH, H2, HID, /*mode=*/1);
  k_head<<<BATCH, 256, 0, stream>>>(bufB, ne_w2, ne_b2, H2, 2, 0, opb, numb, out_num);

  // --- expert select + symbolic ---
  k_expert<<<BATCH, 256, 0, stream>>>(numb, opb, op_w1, op_b1, op_w2, op_b2,
                                      neub, out_neural, out_sym);

  // --- result representation path ---
  k_rank1<<<nblk((size_t)BATCH * H4), 256, 0, stream>>>(neub, re_w1, re_b1, bufA);
  k_gemm_wmma<<<dim3(H2 / 128, BATCH / 128), blk, 0, stream>>>(
      bufA, re2t, re_b2, bufB, BATCH, H2, H4, /*mode=*/1);               // relu, bf16
  k_gemm_wmma<<<dim3(IN_DIM / 128, BATCH / 128), blk, 0, stream>>>(
      bufB, re3t, re_b3, out_repr, BATCH, IN_DIM, H2, /*mode=*/2);       // f32 store
}
